// MultiHeadAttentionLayerGritSparse_19464791785728
// MI455X (gfx1250) — matrix operand order repaired and checked by reading, not compiled, not run on hardware
//
#include <hip/hip_runtime.h>
#include <hip/hip_bf16.h>
#include <math.h>

#define N_NODES 50000
#define N_EDGES 800000
#define IN_DIM  64
#define HEADS   8
#define DHEAD   8
#define HD      64     // HEADS*DHEAD
#define CLAMP_V 5.0f

typedef __attribute__((ext_vector_type(16))) _Float16 v16h;
typedef __attribute__((ext_vector_type(8)))  _Float16 v8h;
typedef __attribute__((ext_vector_type(4)))  _Float16 v4h;
typedef __attribute__((ext_vector_type(8)))  float    v8f;

// ---------------------------------------------------------------------------
// A-matrix loader (global f32 -> f16 frag): 16x32 tile, row stride ld.
// Per CDNA5 ISA 7.12.2: lanes 0-15 hold M=lane, K = {0..7, 16..23};
// lanes 16-31 hold M=lane-16, K = {8..15, 24..31}.
// ---------------------------------------------------------------------------
__device__ __forceinline__ v16h load_a16x32(const float* __restrict__ A,
                                            int row0, int ld, int kb, int lane) {
    int m    = lane & 15;
    int koff = (lane >> 4) << 3;           // 0 or 8
    const float* rp = A + (row0 + m) * ld + kb;
    const float4* p0 = reinterpret_cast<const float4*>(rp + koff);        // K koff..koff+7
    const float4* p1 = reinterpret_cast<const float4*>(rp + 16 + koff);   // K 16+koff..
    float4 x0 = p0[0], x1 = p0[1], x2 = p1[0], x3 = p1[1];
    v16h a;
    a[0]  = (_Float16)x0.x; a[1]  = (_Float16)x0.y; a[2]  = (_Float16)x0.z; a[3]  = (_Float16)x0.w;
    a[4]  = (_Float16)x1.x; a[5]  = (_Float16)x1.y; a[6]  = (_Float16)x1.z; a[7]  = (_Float16)x1.w;
    a[8]  = (_Float16)x2.x; a[9]  = (_Float16)x2.y; a[10] = (_Float16)x2.z; a[11] = (_Float16)x2.w;
    a[12] = (_Float16)x3.x; a[13] = (_Float16)x3.y; a[14] = (_Float16)x3.z; a[15] = (_Float16)x3.w;
    return a;
}

// ---------------------------------------------------------------------------
// A-matrix loader from an LDS-staged f16 tile (16 rows x 64 halfs, row-major).
// Two 16-byte LDS reads per lane per k-step -> ds_load_b128.
// ---------------------------------------------------------------------------
__device__ __forceinline__ v16h load_a16x32_lds(const _Float16* sA, int kb, int lane) {
    int m     = lane & 15;
    int koff8 = lane >> 4;                 // chunk offset 0 or 1 (8 halfs per chunk)
    const v8h* sA8 = reinterpret_cast<const v8h*>(sA);
    v8h c0 = sA8[m * 8 + (kb >> 3) + koff8];        // K kb+koff .. +7
    v8h c1 = sA8[m * 8 + (kb >> 3) + 2 + koff8];    // K kb+16+koff .. +7
    return __builtin_shufflevector(c0, c1, 0, 1, 2, 3, 4, 5, 6, 7,
                                   8, 9, 10, 11, 12, 13, 14, 15);
}

// ---------------------------------------------------------------------------
// B-matrix loader: 32x16 f16 tile (K x N) from row-major f32 weight
// (row stride ldb, columns col0..col0+15). Lanes 0-15 hold N=lane, K={0..7,16..23};
// lanes 16-31 hold K={8..15,24..31}. Weights are tiny and L0/L2 resident.
// ---------------------------------------------------------------------------
__device__ __forceinline__ v16h load_b32x16(const float* __restrict__ W,
                                            int ldb, int kb, int col0, int lane) {
    int nn    = lane & 15;
    int kbase = kb + ((lane >> 4) << 3);   // +0 or +8
    v16h b;
#pragma unroll
    for (int i = 0; i < 16; ++i)
        b[i] = (_Float16)W[(kbase + (i & 7) + ((i >> 3) << 4)) * ldb + col0 + nn];
    return b;
}

__device__ __forceinline__ v8f wmma_f16(v16h a, v16h b, v8f c) {
    return __builtin_amdgcn_wmma_f32_16x16x32_f16(false, a, false, b,
                                                  (short)0, c, false, false);
}

// ---------------------------------------------------------------------------
// Kernel 1: node projections Q/K/V = x @ W + b   (f16 WMMA, f32 accumulate)
// grid = (N/16, 12): y 0-3 -> Q col tiles, 4-7 -> K, 8-11 -> V. block = 32.
// ---------------------------------------------------------------------------
__global__ void node_proj_kernel(const float* __restrict__ x,
                                 const float* __restrict__ WQ, const float* __restrict__ bQ,
                                 const float* __restrict__ WK, const float* __restrict__ bK,
                                 const float* __restrict__ WV, const float* __restrict__ bV,
                                 float* __restrict__ Qh, float* __restrict__ Kh,
                                 float* __restrict__ Vh) {
    int lane  = threadIdx.x & 31;
    int row0  = blockIdx.x << 4;
    int j     = blockIdx.y;
    int which = j >> 2;                 // 0=Q 1=K 2=V
    int col0  = (j & 3) << 4;
    const float* W  = (which == 0) ? WQ : (which == 1) ? WK : WV;
    const float* bb = (which == 0) ? bQ : (which == 1) ? bK : bV;
    float* out = (which == 0) ? Qh : (which == 1) ? Kh : Vh;

    v8f acc = {};
    acc = wmma_f16(load_a16x32(x, row0, IN_DIM, 0,  lane),
                   load_b32x16(W, HD, 0,  col0, lane), acc);
    acc = wmma_f16(load_a16x32(x, row0, IN_DIM, 32, lane),
                   load_b32x16(W, HD, 32, col0, lane), acc);

    int nn   = lane & 15;
    int half = lane >> 4;
    float bias = bb[col0 + nn];
#pragma unroll
    for (int r = 0; r < 8; ++r)
        out[(row0 + r + half * 8) * HD + col0 + nn] = acc[r] + bias;
}

// ---------------------------------------------------------------------------
// Kernel 2: per-edge projection + score + denom accumulation.
// grid = (E/16), block = 256 (8 waves; wave id == head).
// Stage the shared 16x64 edge_attr tile once per block into LDS as f16
// (coalesced 4KB read, 2KB LDS), then each wave computes its head's
// 16-edge x 16-col tile of E = edge_attr @ WE + bE, where for head h
// cols [16h,16h+8) = E_w and [16h+8,16h+16) = E_b.
// score = signed_sqrt((K[src]+Q[dst])*E_w) + E_b -> written to wE,
// logit = sum_d score*Aw -> exp(logit-5) atomically summed into denom[dst,h].
// ---------------------------------------------------------------------------
__global__ void edge_score_kernel(const float* __restrict__ edge_attr,
                                  const int*   __restrict__ eidx,   // [2*E]
                                  const float* __restrict__ WE, const float* __restrict__ bE,
                                  const float* __restrict__ Qh, const float* __restrict__ Kh,
                                  const float* __restrict__ Aw,    // [DHEAD][HEADS]
                                  float* __restrict__ wE_out,      // [E][HD]
                                  float* __restrict__ denom) {     // [N][HEADS]
    __shared__ _Float16 sA[16 * IN_DIM];   // 2KB staged f16 A tile

    int t    = threadIdx.x;                // 0..255
    int lane = t & 31;
    int h    = t >> 5;                     // wave id == head
    int row0 = blockIdx.x << 4;            // edge tile base
    int col0 = h << 4;

    // cooperative stage: one float4 per thread, convert to 4 halfs, ds_store_b64
    {
        const float4* src4 = reinterpret_cast<const float4*>(edge_attr + row0 * IN_DIM);
        float4 v = src4[t];
        v4h hlf;
        hlf[0] = (_Float16)v.x; hlf[1] = (_Float16)v.y;
        hlf[2] = (_Float16)v.z; hlf[3] = (_Float16)v.w;
        reinterpret_cast<v4h*>(sA)[t] = hlf;
    }
    __syncthreads();

    v8f acc = {};
    acc = wmma_f16(load_a16x32_lds(sA, 0,  lane),
                   load_b32x16(WE, 2 * HD, 0,  col0, lane), acc);
    acc = wmma_f16(load_a16x32_lds(sA, 32, lane),
                   load_b32x16(WE, 2 * HD, 32, col0, lane), acc);

    int nn   = lane & 15;
    int half = lane >> 4;
    int d    = lane & 7;
    int hd   = (h << 3) + d;
    float bias = bE[col0 + nn];
    float aw   = Aw[d * HEADS + h];

#pragma unroll
    for (int r = 0; r < 8; ++r) {
        int e   = row0 + r + half * 8;
        int src = eidx[e];
        int dst = eidx[N_EDGES + e];
        float own     = acc[r] + bias;           // lane<8 of group: E_w ; lane>=8: E_b
        float partner = __shfl_xor(own, 8, 32);
        float ew = (lane & 8) ? partner : own;
        float eb = (lane & 8) ? own : partner;

        float kq = Kh[src * HD + hd] + Qh[dst * HD + hd];
        float s  = kq * ew;
        s = sqrtf(fmaxf(s, 0.0f)) - sqrtf(fmaxf(-s, 0.0f));
        s += eb;

        if (!(lane & 8))                          // lanes 0-7 / 16-23 own the output
            wE_out[(size_t)e * HD + hd] = s;

        // logit = sum over d of s * Aw[d,h]; groups of 8 lanes hold duplicates
        float p = s * aw;
        p += __shfl_xor(p, 1, 32);
        p += __shfl_xor(p, 2, 32);
        p += __shfl_xor(p, 4, 32);
        float logit = fminf(fmaxf(p, -CLAMP_V), CLAMP_V);
        float ex    = __expf(logit - CLAMP_V);    // constant shift: softmax-invariant
        if ((lane & 15) == 0)                     // one lane per edge-half
            atomicAdd(&denom[dst * HEADS + h], ex);
    }
}

// ---------------------------------------------------------------------------
// Kernel 3: per (edge, head) aggregation: attn = exp(logit-5)/denom,
// wV[dst] += V[src]*attn ; rowV[dst] += score*attn.
// ---------------------------------------------------------------------------
__global__ void edge_aggregate_kernel(const float* __restrict__ wE_in,   // [E][HD]
                                      const int*   __restrict__ eidx,
                                      const float* __restrict__ Vh,      // [N][HD]
                                      const float* __restrict__ Aw,
                                      const float* __restrict__ denom,   // [N][HEADS]
                                      float* __restrict__ wV,            // [N][HD]
                                      float* __restrict__ rowV) {        // [N][HD]
    int t = blockIdx.x * blockDim.x + threadIdx.x;
    if (t >= N_EDGES * HEADS) return;
    int e = t >> 3;
    int h = t & 7;

    const float4* sc4 = reinterpret_cast<const float4*>(wE_in + (size_t)e * HD + h * DHEAD);
    float4 s0 = sc4[0], s1 = sc4[1];
    float s[8] = { s0.x, s0.y, s0.z, s0.w, s1.x, s1.y, s1.z, s1.w };

    float logit = 0.0f;
#pragma unroll
    for (int d = 0; d < 8; ++d) logit += s[d] * Aw[d * HEADS + h];
    logit = fminf(fmaxf(logit, -CLAMP_V), CLAMP_V);
    float ex = __expf(logit - CLAMP_V);

    int src = eidx[e];
    int dst = eidx[N_EDGES + e];
    float attn = ex / (denom[dst * HEADS + h] + 1e-16f);

    const float* v = Vh + src * HD + h * DHEAD;
    float* ov = wV   + dst * HD + h * DHEAD;
    float* rv = rowV + dst * HD + h * DHEAD;
#pragma unroll
    for (int d = 0; d < 8; ++d) atomicAdd(&ov[d], v[d] * attn);
#pragma unroll
    for (int d = 0; d < 8; ++d) atomicAdd(&rv[d], s[d] * attn);
}

// ---------------------------------------------------------------------------
// Kernel 4: wV[n,h,c] += sum_d rowV[n,h,d] * VeRow[d,h,c]
// ---------------------------------------------------------------------------
__global__ void node_finalize_kernel(const float* __restrict__ rowV,
                                     const float* __restrict__ VeRow,  // [DHEAD][HEADS][DHEAD]
                                     float* __restrict__ wV) {
    int t = blockIdx.x * blockDim.x + threadIdx.x;
    if (t >= N_NODES * HD) return;
    int n  = t >> 6;
    int hc = t & 63;
    int h  = hc >> 3;
    int c  = hc & 7;
    float acc = wV[t];
    const float* rv = rowV + n * HD + h * DHEAD;
#pragma unroll
    for (int d = 0; d < 8; ++d)
        acc += rv[d] * VeRow[d * HD + h * DHEAD + c];
    wV[t] = acc;
}

// ---------------------------------------------------------------------------
extern "C" void kernel_launch(void* const* d_in, const int* in_sizes, int n_in,
                              void* d_out, int out_size, void* d_ws, size_t ws_size,
                              hipStream_t stream) {
    const float* x         = (const float*)d_in[0];
    const float* edge_attr = (const float*)d_in[1];
    const int*   edge_idx  = (const int*)  d_in[2];
    const float* WQ = (const float*)d_in[3];
    const float* bQ = (const float*)d_in[4];
    const float* WK = (const float*)d_in[5];
    const float* bK = (const float*)d_in[6];
    const float* WE = (const float*)d_in[7];
    const float* bE = (const float*)d_in[8];
    const float* WV = (const float*)d_in[9];
    const float* bV = (const float*)d_in[10];
    const float* Aw = (const float*)d_in[11];
    const float* VeRow = (const float*)d_in[12];

    float* out_wV = (float*)d_out;                         // [N][HD]
    float* out_wE = (float*)d_out + (size_t)N_NODES * HD;  // [E][HD]

    // workspace layout (floats)
    float* Qh    = (float*)d_ws;
    float* Kh    = Qh + (size_t)N_NODES * HD;
    float* Vh    = Kh + (size_t)N_NODES * HD;
    float* denom = Vh + (size_t)N_NODES * HD;              // [N][HEADS]
    float* rowV  = denom + (size_t)N_NODES * HEADS;        // [N][HD]

    // zero the atomically-accumulated buffers (graph-capturable memset nodes)
    hipMemsetAsync(out_wV, 0, (size_t)N_NODES * HD * sizeof(float), stream);
    hipMemsetAsync(denom,  0, (size_t)N_NODES * HEADS * sizeof(float), stream);
    hipMemsetAsync(rowV,   0, (size_t)N_NODES * HD * sizeof(float), stream);

    // 1) node projections (WMMA)
    node_proj_kernel<<<dim3(N_NODES / 16, 12), 32, 0, stream>>>(
        x, WQ, bQ, WK, bK, WV, bV, Qh, Kh, Vh);

    // 2) edge projection + score + denom (WMMA, LDS-staged A tile)
    edge_score_kernel<<<N_EDGES / 16, 256, 0, stream>>>(
        edge_attr, edge_idx, WE, bE, Qh, Kh, Aw, out_wE, denom);

    // 3) per-edge aggregation into wV / rowV
    {
        int total = N_EDGES * HEADS;
        edge_aggregate_kernel<<<(total + 255) / 256, 256, 0, stream>>>(
            out_wE, edge_idx, Vh, Aw, denom, out_wV, rowV);
    }

    // 4) wV += rowV @ VeRow (per head 8x8)
    {
        int total = N_NODES * HD;
        node_finalize_kernel<<<(total + 255) / 256, 256, 0, stream>>>(
            rowV, VeRow, out_wV);
    }
}
